// Interaction_43456479101762
// MI455X (gfx1250) — matrix operand order repaired and checked by reading, not compiled, not run on hardware
//
#include <hip/hip_runtime.h>
#include <hip/hip_bf16.h>

// ---------------------------------------------------------------------------
// Types for CDNA5 WMMA (wave32)
// ---------------------------------------------------------------------------
typedef __attribute__((ext_vector_type(16))) __bf16          v16bf;
typedef __attribute__((ext_vector_type(8)))  float           v8f;
typedef __attribute__((ext_vector_type(16))) unsigned short  u16x16;
typedef __attribute__((ext_vector_type(4)))  unsigned short  u16x4;
typedef __attribute__((ext_vector_type(4)))  float           f32x4;

// round-to-nearest-even f32 -> bf16 bits
__device__ __forceinline__ unsigned short f2bfbits(float f) {
    unsigned u = __float_as_uint(f);
    unsigned r = u + 0x7FFFu + ((u >> 16) & 1u);
    return (unsigned short)(r >> 16);
}

// K-swizzle (involution): inside each 32-wide K block, swap [8..15] <-> [16..23].
// A-side bf16 tensors are STORED swizzled so a WMMA A fragment is one
// contiguous 32-byte run per lane (see fragment layout notes below).
__device__ __forceinline__ int  kswz (int  k) {
    int q = (k >> 3) & 3;
    return k + ((q == 1) ? 8 : (q == 2) ? -8 : 0);
}
__device__ __forceinline__ long kswzl(long k) {
    int q = (int)(k >> 3) & 3;
    return k + ((q == 1) ? 8 : (q == 2) ? -8 : 0);
}

// ---------------------------------------------------------------------------
// WMMA fragment loaders (layouts per CDNA5 ISA 7.12.2)
// A (16x32 bf16, swizzled storage):
//   lane<16  -> row=lane,    elements = swizzled [k0 .. k0+15]  (= K 0-7,16-23)
//   lane>=16 -> row=lane-16, elements = swizzled [k0+16..k0+31] (= K 8-15,24-31)
// B (32x16 bf16, natural N-major/K-contiguous storage):
//   lane<16  -> col=lane,    K = k0+0..7,16..23 ; lane>=16 -> col=lane-16, K = k0+8..15,24..31
// C/D (16x16 f32): vgpr r, lane -> row = r + 8*(lane>=16), col = lane&15
// Each A fragment = ONE contiguous 32-byte load (2 adjacent b128s from one
// vector load -> no register-coalescing movs).
// ---------------------------------------------------------------------------
__device__ __forceinline__ v16bf load_frag32B(const unsigned short* p) {
    return __builtin_bit_cast(v16bf, *(const u16x16*)p);
}

// A stored row-major MxK (bf16 bits, K-swizzled), global
__device__ __forceinline__ v16bf load_a_frag(const unsigned short* __restrict__ A,
                                             long row0, int k0, int lda) {
    int lane = threadIdx.x & 31;
    return load_frag32B(A + (row0 + (lane & 15)) * (long)lda + k0 + ((lane >> 4) << 4));
}

// A fragment out of bf16 LDS tile (16 x ld, K-swizzled)
__device__ __forceinline__ v16bf load_a_frag_lds(const unsigned short* __restrict__ P,
                                                 int k0, int ld) {
    int lane = threadIdx.x & 31;
    return load_frag32B(P + (lane & 15) * ld + k0 + ((lane >> 4) << 4));
}

// B supplied as Bt: N-major, K-contiguous (i.e. Bt[n*ldb + k] == B[k, n])
__device__ __forceinline__ v16bf load_b_frag(const unsigned short* __restrict__ Bt,
                                             long n0, int k0, int ldb) {
    int lane = threadIdx.x & 31;
    typedef __attribute__((ext_vector_type(8))) unsigned short u16x8;
    const unsigned short* p = Bt + (n0 + (lane & 15)) * (long)ldb + k0 + ((lane >> 4) << 3);
    u16x8 lo = *(const u16x8*)p;
    u16x8 hi = *(const u16x8*)(p + 16);
    return __builtin_bit_cast(v16bf, __builtin_shufflevector(lo, hi, 0, 1, 2, 3, 4, 5, 6, 7, 8, 9, 10, 11, 12, 13, 14, 15));
}

__device__ __forceinline__ float wave_sum(float v) {
#pragma unroll
    for (int o = 16; o > 0; o >>= 1) v += __shfl_xor(v, o, 32);
    return v;
}
__device__ __forceinline__ float wave_max(float v) {
#pragma unroll
    for (int o = 16; o > 0; o >>= 1) v = fmaxf(v, __shfl_xor(v, o, 32));
    return v;
}

// ---------------------------------------------------------------------------
// f32 -> bf16 conversions
// ---------------------------------------------------------------------------
// Elementwise, with optional A-side K-swizzle on the store index.
// n4 = n/4; n and the row length must be multiples of 32 (ours are 512-mult).
__global__ void cvt_bf16_kernel(const float* __restrict__ in,
                                unsigned short* __restrict__ out, long n4, int aswz) {
    long i = (long)blockIdx.x * blockDim.x + threadIdx.x;
    if (i < n4) {
        f32x4 v = ((const f32x4*)in)[i];
        u16x4 o;
#pragma unroll
        for (int k = 0; k < 4; ++k) o[k] = f2bfbits(v[k]);
        long base = i * 4;
        if (aswz) base = kswzl(base);              // moves whole 8-groups; 4-aligned safe
        *(u16x4*)(out + base) = o;
    }
}

// in: R x C row-major  ->  out: C x R row-major (transposed), bf16 bits
__global__ void cvt_bf16_T_kernel(const float* __restrict__ in,
                                  unsigned short* __restrict__ out, int R, int C) {
    long i = (long)blockIdx.x * blockDim.x + threadIdx.x;
    if (i < (long)R * C) {
        int r = (int)(i / C), c = (int)(i % C);
        out[(long)c * R + r] = f2bfbits(in[i]);
    }
}

// ---------------------------------------------------------------------------
// Generic bf16 WMMA GEMM:  C[M,N] = scale * A[M,K] x B[K,N] (+bias) (+relu)
// A: M x K bf16 row-major, K-swizzled.  Bt: N x K bf16 (K-contiguous).
// Explicit 2-deep ping-pong pipeline over K (K/32 even at all call sites).
// Cf (f32, optional), Cb (bf16, optional):
//   tmode==0: row-major; aswz!=0 additionally K-swizzles the column index
//             (output feeds a later GEMM/WMMA as an A operand).
//   tmode==1: per-batch transposed layout (batch, n, l), l = row % batchL
//             (attention V^T input; consumed as B -> no swizzle).
// Block: 256 threads / 8 waves -> 32x256 tile; wave computes 16x64.
// ---------------------------------------------------------------------------
__global__ __launch_bounds__(256) void gemm_bf16_kernel(
    const unsigned short* __restrict__ Ap, const unsigned short* __restrict__ Btp,
    const float* __restrict__ bias, float* __restrict__ Cf,
    unsigned short* __restrict__ Cb, int M, int N, int K,
    int relu, int tmode, int aswz, int batchL, float scale)
{
    int wv   = threadIdx.x >> 5;
    int lane = threadIdx.x & 31;
    long row0 = (long)blockIdx.x * 32 + ((wv >> 2) << 4);
    int  col0 = blockIdx.y * 256 + ((wv & 3) << 6);

    v8f acc[4];
#pragma unroll
    for (int t = 0; t < 4; ++t) acc[t] = {};

    // prologue: two fragment generations in flight
    v16bf aA = load_a_frag(Ap, row0, 0, K);
    v16bf bA[4];
#pragma unroll
    for (int t = 0; t < 4; ++t) bA[t] = load_b_frag(Btp, (long)col0 + t * 16, 0, K);
    v16bf aB = load_a_frag(Ap, row0, 32, K);
    v16bf bB[4];
#pragma unroll
    for (int t = 0; t < 4; ++t) bB[t] = load_b_frag(Btp, (long)col0 + t * 16, 32, K);

    for (int kc = 64; kc + 64 <= K; kc += 64) {
        __builtin_prefetch(Ap + (row0 + (lane & 15)) * (long)K + kc + 64, 0, 3);
        // consume generation A (kc-64), then reload it for kc
#pragma unroll
        for (int t = 0; t < 4; ++t)
            acc[t] = __builtin_amdgcn_wmma_f32_16x16x32_bf16(
                false, aA, false, bA[t], (short)0, acc[t], false, false);
        aA = load_a_frag(Ap, row0, kc, K);
#pragma unroll
        for (int t = 0; t < 4; ++t)
            bA[t] = load_b_frag(Btp, (long)col0 + t * 16, kc, K);
        // consume generation B (kc-32), then reload it for kc+32
#pragma unroll
        for (int t = 0; t < 4; ++t)
            acc[t] = __builtin_amdgcn_wmma_f32_16x16x32_bf16(
                false, aB, false, bB[t], (short)0, acc[t], false, false);
        aB = load_a_frag(Ap, row0, kc + 32, K);
#pragma unroll
        for (int t = 0; t < 4; ++t)
            bB[t] = load_b_frag(Btp, (long)col0 + t * 16, kc + 32, K);
    }
    // tail: last two chunks already resident
#pragma unroll
    for (int t = 0; t < 4; ++t)
        acc[t] = __builtin_amdgcn_wmma_f32_16x16x32_bf16(
            false, aA, false, bA[t], (short)0, acc[t], false, false);
#pragma unroll
    for (int t = 0; t < 4; ++t)
        acc[t] = __builtin_amdgcn_wmma_f32_16x16x32_bf16(
            false, aB, false, bB[t], (short)0, acc[t], false, false);

    int rbase = (lane >> 4) << 3;
    int nlane = lane & 15;
#pragma unroll
    for (int t = 0; t < 4; ++t) {
        int col = col0 + t * 16 + nlane;
        float bv = bias ? bias[col] : 0.0f;
#pragma unroll
        for (int r = 0; r < 8; ++r) {
            long row = row0 + rbase + r;
            float v = acc[t][r] * scale + bv;
            if (relu) v = fmaxf(v, 0.0f);
            if (Cf) Cf[row * (long)N + col] = v;
            if (Cb) {
                unsigned short bb = f2bfbits(v);
                if (tmode == 0) {
                    int colw = aswz ? kswz(col) : col;
                    Cb[row * (long)N + colw] = bb;
                } else {
                    long b_ = row / batchL;
                    int  l_ = (int)(row % batchL);
                    Cb[(b_ * N + col) * (long)batchL + l_] = bb;
                }
            }
        }
    }
}

// ---------------------------------------------------------------------------
// Attention: one workgroup per (batch, 16 query rows). 512 threads = 16 waves.
// LDS: S  = 16 x L scores, fp32 (128 KB)
//      P  = 16 x L exp(s-m), bf16, K-swizzled (64 KB) -- un-normalized
//      ri = 16 fp32 reciprocal row sums
// Phase 1: S = Q K^T / sqrt(Dq) via WMMA (Q preloaded; B rotated through the
//          fully-unrolled K loop -> pure register renaming).
// Phase 2: row max + exp into bf16 P (swizzled store); 1/sum per row (exact
//          fp32 softmax, normalization folded into the phase-3 epilogue).
// Phase 3: O = (P V) * (1/sum_row) via WMMA, 2-deep ping-pong pipeline.
// ---------------------------------------------------------------------------
__global__ __launch_bounds__(512) void attn_kernel(
    const unsigned short* __restrict__ Qb, const unsigned short* __restrict__ Kb,
    const unsigned short* __restrict__ Vtb, float* __restrict__ Out,
    int Ln, int Dq, int Av)
{
    extern __shared__ __align__(32) char smem[];
    float*          S  = (float*)smem;                               // 16*Ln f32
    unsigned short* P  = (unsigned short*)(smem + 16 * Ln * 4);      // 16*Ln bf16
    float*          ri = (float*)(smem + 16 * Ln * 6);               // 16 f32

    int wv   = threadIdx.x >> 5;
    int lane = threadIdx.x & 31;
    int tilesPerRow = Ln >> 4;                     // 128
    int bidx = blockIdx.x / tilesPerRow;
    int l0   = (blockIdx.x % tilesPerRow) << 4;
    long qrow0 = (long)bidx * Ln + l0;
    float sc = rsqrtf((float)Dq);

    int rbase = (lane >> 4) << 3;
    int nlane = lane & 15;

    // ---- Phase 1: scores into LDS (Q preloaded: Dq/32 = 16 fragments) ----
    v16bf qf[16];
#pragma unroll
    for (int kf = 0; kf < 16; ++kf)
        qf[kf] = load_a_frag(Qb, qrow0, kf << 5, Dq);

    int ntPerWave = tilesPerRow / 16;              // 8 column tiles per wave
    for (int i = 0; i < ntPerWave; ++i) {
        int col0 = (wv * ntPerWave + i) << 4;
        long krow = (long)bidx * Ln + col0;
        v8f acc = {};
        v16bf bcur = load_b_frag(Kb, krow, 0, Dq);
#pragma unroll
        for (int kf = 0; kf < 16; ++kf) {
            v16bf bnext = (kf < 15) ? load_b_frag(Kb, krow, (kf + 1) << 5, Dq) : bcur;
            acc = __builtin_amdgcn_wmma_f32_16x16x32_bf16(
                false, qf[kf], false, bcur, (short)0, acc, false, false);
            bcur = bnext;
        }
#pragma unroll
        for (int r = 0; r < 8; ++r)
            S[(rbase + r) * Ln + col0 + nlane] = acc[r] * sc;
    }
    __syncthreads();

    // ---- Phase 2: softmax numerator -> bf16 P (K-swizzled), 1/sum per row -
    {
        float*          rowS = S + wv * Ln;
        unsigned short* rowP = P + wv * Ln;
        float m = -3.0e38f;
        for (int j = lane; j < Ln; j += 32) m = fmaxf(m, rowS[j]);
        m = wave_max(m);
        float s = 0.0f;
        for (int j = lane; j < Ln; j += 32) {
            float e = __expf(rowS[j] - m);
            s += e;
            rowP[kswz(j)] = f2bfbits(e);
        }
        s = wave_sum(s);
        if (lane == 0) ri[wv] = 1.0f / s;
    }
    __syncthreads();

    // ---- Phase 3: O = (P x V) * ri[row], 2-deep ping-pong ---------------
    {
        v8f acc[2];
#pragma unroll
        for (int t = 0; t < 2; ++t) acc[t] = {};
        int  colbase = wv << 5;                    // 2 x 16 output cols per wave
        long vrow0   = (long)bidx * Av + colbase;

        v16bf aA = load_a_frag_lds(P, 0, Ln);
        v16bf bA[2];
#pragma unroll
        for (int t = 0; t < 2; ++t) bA[t] = load_b_frag(Vtb, vrow0 + t * 16, 0, Ln);
        v16bf aB = load_a_frag_lds(P, 32, Ln);
        v16bf bB[2];
#pragma unroll
        for (int t = 0; t < 2; ++t) bB[t] = load_b_frag(Vtb, vrow0 + t * 16, 32, Ln);

        for (int kc = 64; kc + 64 <= Ln; kc += 64) {
#pragma unroll
            for (int t = 0; t < 2; ++t)
                acc[t] = __builtin_amdgcn_wmma_f32_16x16x32_bf16(
                    false, aA, false, bA[t], (short)0, acc[t], false, false);
            aA = load_a_frag_lds(P, kc, Ln);
#pragma unroll
            for (int t = 0; t < 2; ++t)
                bA[t] = load_b_frag(Vtb, vrow0 + t * 16, kc, Ln);
#pragma unroll
            for (int t = 0; t < 2; ++t)
                acc[t] = __builtin_amdgcn_wmma_f32_16x16x32_bf16(
                    false, aB, false, bB[t], (short)0, acc[t], false, false);
            aB = load_a_frag_lds(P, kc + 32, Ln);
#pragma unroll
            for (int t = 0; t < 2; ++t)
                bB[t] = load_b_frag(Vtb, vrow0 + t * 16, kc + 32, Ln);
        }
#pragma unroll
        for (int t = 0; t < 2; ++t)
            acc[t] = __builtin_amdgcn_wmma_f32_16x16x32_bf16(
                false, aA, false, bA[t], (short)0, acc[t], false, false);
#pragma unroll
        for (int t = 0; t < 2; ++t)
            acc[t] = __builtin_amdgcn_wmma_f32_16x16x32_bf16(
                false, aB, false, bB[t], (short)0, acc[t], false, false);

#pragma unroll
        for (int t = 0; t < 2; ++t) {
#pragma unroll
            for (int r = 0; r < 8; ++r) {
                float inv = ri[rbase + r];
                Out[(qrow0 + rbase + r) * (long)Av + colbase + t * 16 + nlane] =
                    acc[t][r] * inv;
            }
        }
    }
}

// ---------------------------------------------------------------------------
// out = a * ((X+Y) - mean) / (std_ddof1 + eps) + b   (per 512-elem row)
// One wave per row, b128 I/O; bf16 copy (K-swizzled, feeds GEMM A) if Ob != 0.
// ---------------------------------------------------------------------------
template <int NCOLS>
__global__ __launch_bounds__(256) void add_ln_kernel(
    const float* __restrict__ X, const float* __restrict__ Y,
    const float* __restrict__ ga, const float* __restrict__ gb,
    float* __restrict__ Of, unsigned short* __restrict__ Ob, long Mrows)
{
    long row = (long)blockIdx.x * 8 + (threadIdx.x >> 5);
    if (row >= Mrows) return;
    int lane = threadIdx.x & 31;

    constexpr int CH = NCOLS / 128;                // float4 chunks per lane
    f32x4 h[CH];
    float s = 0.0f, s2 = 0.0f;
#pragma unroll
    for (int c = 0; c < CH; ++c) {
        long j = row * NCOLS + c * 128 + lane * 4;
        f32x4 xv = *(const f32x4*)(X + j);
        f32x4 yv = *(const f32x4*)(Y + j);
        f32x4 v  = xv + yv;
        h[c] = v;
#pragma unroll
        for (int k = 0; k < 4; ++k) { s += v[k]; s2 += v[k] * v[k]; }
    }
    s  = wave_sum(s);
    s2 = wave_sum(s2);
    float mean = s / NCOLS;
    float var  = (s2 - s * mean) / (NCOLS - 1);    // Bessel (ddof=1)
    float denom = sqrtf(fmaxf(var, 0.0f)) + 1e-6f;
    float aa = ga[0], bb = gb[0];
    float inv = aa / denom;
#pragma unroll
    for (int c = 0; c < CH; ++c) {
        long j = row * NCOLS + c * 128 + lane * 4;
        f32x4 y;
        u16x4 yb;
#pragma unroll
        for (int k = 0; k < 4; ++k) {
            y[k]  = (h[c][k] - mean) * inv + bb;
            yb[k] = f2bfbits(y[k]);
        }
        *(f32x4*)(Of + j) = y;
        if (Ob) *(u16x4*)(Ob + kswzl(j)) = yb;     // 4-aligned; swizzle moves 8-groups
    }
}

// ---------------------------------------------------------------------------
// Orchestration
// ---------------------------------------------------------------------------
extern "C" void kernel_launch(void* const* d_in, const int* in_sizes, int n_in,
                              void* d_out, int out_size, void* d_ws, size_t ws_size,
                              hipStream_t stream) {
    const float* x   = (const float*)d_in[0];
    const float* Wq  = (const float*)d_in[1];
    const float* Wk  = (const float*)d_in[2];
    const float* Wv  = (const float*)d_in[3];
    const float* W1  = (const float*)d_in[4];
    const float* b1  = (const float*)d_in[5];
    const float* W2  = (const float*)d_in[6];
    const float* b2  = (const float*)d_in[7];
    const float* a1  = (const float*)d_in[8];
    const float* bb1 = (const float*)d_in[9];
    const float* a2  = (const float*)d_in[10];
    const float* bb2 = (const float*)d_in[11];

    const int  B = 32, L = 2048, A = 512, DQ = 512, H = 2048;
    const long M = (long)B * L;                     // 65536 rows

    // workspace carve-up
    char* ws = (char*)d_ws;
    size_t off = 0;
    auto carve = [&](size_t bytes) {
        void* p = ws + off;
        off += (bytes + 255) & ~(size_t)255;
        return p;
    };
    unsigned short* xb   = (unsigned short*)carve(M * A * 2);   // K-swizzled (A-side)
    unsigned short* qb   = (unsigned short*)carve(M * DQ * 2);  // K-swizzled (A-side)
    unsigned short* kb   = (unsigned short*)carve(M * DQ * 2);  // natural (B-side)
    unsigned short* vtb  = (unsigned short*)carve(M * A * 2);   // (b, a, l)  (B-side)
    unsigned short* wqt  = (unsigned short*)carve((size_t)A * DQ * 2);
    unsigned short* wkt  = (unsigned short*)carve((size_t)A * DQ * 2);
    unsigned short* wvt  = (unsigned short*)carve((size_t)A * A * 2);
    unsigned short* w1t  = (unsigned short*)carve((size_t)A * H * 2);
    unsigned short* w2t  = (unsigned short*)carve((size_t)H * A * 2);
    float*          attn = (float*)carve(M * A * 4);
    float*          n1f  = (float*)carve(M * A * 4);
    unsigned short* n1b  = (unsigned short*)carve(M * A * 2);   // K-swizzled (A-side)
    unsigned short* ffnb = (unsigned short*)carve(M * H * 2);   // K-swizzled (A-side)
    float*          ffn2 = attn;                    // reuse after n1 is built

    // 1) conversions (x elementwise+swizzle; weights transposed, K-contiguous)
    {
        long n4 = M * A / 4;
        cvt_bf16_kernel<<<dim3((unsigned)((n4 + 255) / 256)), dim3(256), 0, stream>>>(
            x, xb, n4, /*aswz=*/1);
        cvt_bf16_T_kernel<<<dim3((A * DQ + 255) / 256), dim3(256), 0, stream>>>(Wq, wqt, A, DQ);
        cvt_bf16_T_kernel<<<dim3((A * DQ + 255) / 256), dim3(256), 0, stream>>>(Wk, wkt, A, DQ);
        cvt_bf16_T_kernel<<<dim3((A * A  + 255) / 256), dim3(256), 0, stream>>>(Wv, wvt, A, A);
        cvt_bf16_T_kernel<<<dim3((A * H  + 255) / 256), dim3(256), 0, stream>>>(W1, w1t, A, H);
        cvt_bf16_T_kernel<<<dim3((H * A  + 255) / 256), dim3(256), 0, stream>>>(W2, w2t, H, A);
    }

    // 2) QKV projections (WMMA). Q is consumed as an A operand -> swizzled;
    //    K is consumed as a B operand -> natural; V -> per-batch transposed.
    {
        dim3 g((unsigned)(M / 32), DQ / 256), blk(256);
        gemm_bf16_kernel<<<g, blk, 0, stream>>>(xb, wqt, nullptr, nullptr, qb,
                                                (int)M, DQ, A, 0, 0, /*aswz=*/1, L, 1.0f);
        gemm_bf16_kernel<<<g, blk, 0, stream>>>(xb, wkt, nullptr, nullptr, kb,
                                                (int)M, DQ, A, 0, 0, /*aswz=*/0, L, 1.0f);
        gemm_bf16_kernel<<<g, blk, 0, stream>>>(xb, wvt, nullptr, nullptr, vtb,
                                                (int)M, A, A, 0, /*tmode=*/1, 0, L, 1.0f);
    }

    // 3) attention (192 KB dynamic LDS per workgroup: f32 S + bf16 P + 1/sum)
    {
        dim3 g((unsigned)(B * (L / 16))), blk(512);
        size_t shmem = (size_t)16 * L * 6 + 64;
        attn_kernel<<<g, blk, shmem, stream>>>(qb, kb, vtb, attn, L, DQ, A);
    }

    // 4) residual + LayerNorm 1  (f32 out for residual, swizzled bf16 for FFN)
    add_ln_kernel<512><<<dim3((unsigned)(M / 8)), dim3(256), 0, stream>>>(
        x, attn, a1, bb1, n1f, n1b, M);

    // 5) FFN (ffnb feeds FFN2 as its A operand -> swizzled store)
    {
        dim3 g1((unsigned)(M / 32), H / 256), blk(256);
        gemm_bf16_kernel<<<g1, blk, 0, stream>>>(n1b, w1t, b1, nullptr, ffnb,
                                                 (int)M, H, A, /*relu=*/1, 0, /*aswz=*/1, L, 1.0f);
        dim3 g2((unsigned)(M / 32), A / 256);
        gemm_bf16_kernel<<<g2, blk, 0, stream>>>(ffnb, w2t, b2, ffn2, nullptr,
                                                 (int)M, A, H, 0, 0, 0, L, 1.0f);
    }

    // 6) residual + LayerNorm 2 -> final output
    add_ln_kernel<512><<<dim3((unsigned)(M / 8)), dim3(256), 0, stream>>>(
        n1f, ffn2, a2, bb2, (float*)d_out, nullptr, M);

    (void)in_sizes; (void)n_in; (void)out_size; (void)ws_size;
}
